// GCNN_53953379172478
// MI455X (gfx1250) — matrix operand order repaired and checked by reading, not compiled, not run on hardware
//
#include <hip/hip_runtime.h>
#include <hip/hip_bf16.h>

typedef __attribute__((ext_vector_type(8)))  _Float16 v8h;
typedef __attribute__((ext_vector_type(16))) _Float16 v16h;
typedef __attribute__((ext_vector_type(8)))  float    v8f;

#define HWD   28
#define IMG   784          // 28*28
#define NPOS  401408       // 512*784
#define C1OUT 128          // G*O1
#define PELEM 25088        // 32*784 pooled elements per image
#define NCLS  10

// ---------------- kernel 1: rotated first-layer filters (fp32 bilinear) -------------
__global__ void rot_filters(const float* __restrict__ base, float* __restrict__ w1f) {
  int e = blockIdx.x * blockDim.x + threadIdx.x;
  if (e >= 8 * 16 * 9) return;
  int k = e / 144, r = e % 144, o = r / 9, q = r % 9, iy = q / 3, ix = q % 3;
  const float coords[3] = {-2.f / 3.f, 0.f, 2.f / 3.f};
  float xs = coords[ix], ys = coords[iy];
  float th = 2.f * 3.14159265358979323846f * (float)k / 8.f;
  float c = cosf(th), s = sinf(th);
  float gx = c * xs - s * ys, gy = s * xs + c * ys;
  float px = ((gx + 1.f) * 3.f - 1.f) * 0.5f;
  float py = ((gy + 1.f) * 3.f - 1.f) * 0.5f;
  float x0 = floorf(px), y0 = floorf(py);
  float wx = px - x0, wy = py - y0;
  int x0i = (int)x0, y0i = (int)y0;
  const float* bo = base + o * 9;
  auto g = [&](int yi, int xi) -> float {
    bool v = (yi >= 0) && (yi < 3) && (xi >= 0) && (xi < 3);
    int yc = yi < 0 ? 0 : (yi > 2 ? 2 : yi);
    int xc = xi < 0 ? 0 : (xi > 2 ? 2 : xi);
    return v ? bo[yc * 3 + xc] : 0.f;
  };
  float val = g(y0i, x0i) * (1.f - wx) * (1.f - wy)
            + g(y0i, x0i + 1) * wx * (1.f - wy)
            + g(y0i + 1, x0i) * (1.f - wx) * wy
            + g(y0i + 1, x0i + 1) * wx * wy;
  w1f[e] = val;  // e == (k*16+o)*9 + q  == [oc][tap]
}

// --------------- kernel 2: pack group-circulant conv2 weights -> f16 [tap][oc][ic] ---
__global__ void pack_w2(const float* __restrict__ w2, _Float16* __restrict__ wp) {
  int e = blockIdx.x * 256 + threadIdx.x;  // 9*256*128 = 294912
  if (e >= 294912) return;
  int t = e / 32768, r = e % 32768, oc = r / 128, ic = r % 128;
  int g = oc >> 5, o = oc & 31, c2 = ic >> 3, hh = ic & 7;
  int gg = (g - hh + 8) & 7;
  float v = w2[(((o * 16 + c2) * 8 + gg) * 9) + t];
  wp[e] = (_Float16)v;
}

// --------------- kernel 3: conv1 (1->128 ch) + relu, f16 channel-major output --------
__global__ void conv1_relu(const float* __restrict__ x, const float* __restrict__ w1f,
                           _Float16* __restrict__ h) {
  const int oc  = blockIdx.y;
  const int pos = blockIdx.x * 256 + threadIdx.x;
  float w[9];
#pragma unroll
  for (int t = 0; t < 9; ++t) w[t] = w1f[oc * 9 + t];
  const int b = pos / IMG, rem = pos % IMG, y = rem / HWD, xx = rem % HWD;
  const float* xi = x + (size_t)b * IMG;
  float acc = 0.f;
#pragma unroll
  for (int dy = -1; dy <= 1; ++dy)
#pragma unroll
    for (int dx = -1; dx <= 1; ++dx) {
      int yy = y + dy, xc = xx + dx;
      if (yy >= 0 && yy < HWD && xc >= 0 && xc < HWD)
        acc += w[(dy + 1) * 3 + (dx + 1)] * xi[yy * HWD + xc];
    }
  h[(size_t)oc * NPOS + pos] = (_Float16)fmaxf(acc, 0.f);
}

// --------------- kernel 4: conv2 as WMMA implicit GEMM + relu + group pooling --------
// GEMM: M=256 (oc), K=1152 (9 taps x 128 ic), N=401408 (b,y,x)
// Block tile 128M x 128N, 8 waves each owning 64M x 32N (8 WMMA accumulators).
// All fragment ds-loads are issued before the WMMA chain so one s_wait_dscnt
// covers a back-to-back run of 8 independent v_wmma ops.
__launch_bounds__(256)
__global__ void conv2_wmma_pool(const _Float16* __restrict__ h,
                                const _Float16* __restrict__ wp,
                                float* __restrict__ p) {
  __shared__ __align__(16) _Float16 As[2][128][48];  // [m][k], padded rows (96B stride)
  __shared__ __align__(16) _Float16 Bs[2][128][48];  // [n][k]

  const int tid  = threadIdx.x;
  const int lane = tid & 31, wave = tid >> 5;
  const int wm = wave >> 2, wn = wave & 3;
  const int oc_base = blockIdx.y * 128;
  const int n_base  = blockIdx.x * 128;

  // cooperative loader indices
  const int mA = tid >> 1, kA = (tid & 1) * 16;          // A: 128 rows x 32 k
  const int nB = tid & 127, kq = tid >> 7;               // B: 128 n x 32 k (16 k/thread)
  const int posB = n_base + nB;
  const int bB = posB / IMG, remB = posB % IMG;
  const int yB = remB / HWD, xB = remB % HWD;

  // fragment indices (CDNA5 wave32 WMMA layouts)
  const int nf  = lane & 15;
  const int kbA = (lane < 16) ? 0 : 16;
  const int kbB = (lane < 16) ? 0 : 16;
  const int mrow = wm * 64;

  v8f acc[2][4] = {};
  v8h ra0, ra1, rb0, rb1;

  auto load_global = [&](int ks) {
    const int t = ks >> 2, icb = (ks & 3) * 32;
    const v8h* ap = (const v8h*)(wp + ((size_t)(t * 256 + oc_base + mA)) * 128 + icb + kA);
    ra0 = ap[0];
    ra1 = ap[1];
    const int dy = t / 3 - 1, dx = t % 3 - 1;
    const int yy = yB + dy, xc = xB + dx;
    const bool valid = (yy >= 0) && (yy < HWD) && (xc >= 0) && (xc < HWD);
    const long pb = (long)bB * IMG + yy * HWD + xc;
    const int ic0 = icb + kq * 16;
#pragma unroll
    for (int j = 0; j < 8; ++j)
      rb0[j] = valid ? h[(long)(ic0 + j) * NPOS + pb] : (_Float16)0.f;
#pragma unroll
    for (int j = 0; j < 8; ++j)
      rb1[j] = valid ? h[(long)(ic0 + 8 + j) * NPOS + pb] : (_Float16)0.f;
  };
  auto store_lds = [&](int buf) {
    *(v8h*)&As[buf][mA][kA]          = ra0;
    *(v8h*)&As[buf][mA][kA + 8]      = ra1;
    *(v8h*)&Bs[buf][nB][kq * 16]     = rb0;
    *(v8h*)&Bs[buf][nB][kq * 16 + 8] = rb1;
  };

  load_global(0);
  for (int ks = 0; ks < 36; ++ks) {
    const int cur = ks & 1;
    store_lds(cur);
    __syncthreads();
    if (ks < 35) {
      load_global(ks + 1);
      if (ks < 34) {  // prefetch weight tile two steps ahead -> global_prefetch_b8
        const int t2 = (ks + 2) >> 2, icb2 = ((ks + 2) & 3) * 32;
        __builtin_prefetch(wp + ((size_t)(t2 * 256 + oc_base + mA)) * 128 + icb2 + kA, 0, 1);
      }
    }

    // ---- load ALL fragments first (12 ds_load_b128), then dense WMMA chain ----
    v16h bf[2], af[4];
#pragma unroll
    for (int q = 0; q < 2; ++q) {
      const int n = wn * 32 + q * 16 + nf;
      v8h b0 = *(const v8h*)&Bs[cur][n][kbB];
      v8h b1 = *(const v8h*)&Bs[cur][n][kbB + 8];
      bf[q] = __builtin_shufflevector(b0, b1, 0, 1, 2, 3, 4, 5, 6, 7,
                                      8, 9, 10, 11, 12, 13, 14, 15);
    }
#pragma unroll
    for (int s = 0; s < 4; ++s) {
      const int m = mrow + s * 16 + nf;
      v8h a0 = *(const v8h*)&As[cur][m][kbA];
      v8h a1 = *(const v8h*)&As[cur][m][kbA + 8];
      af[s] = __builtin_shufflevector(a0, a1, 0, 1, 2, 3, 4, 5, 6, 7,
                                      8, 9, 10, 11, 12, 13, 14, 15);
    }
#pragma unroll
    for (int q = 0; q < 2; ++q)
#pragma unroll
      for (int s = 0; s < 4; ++s)
        acc[q][s] = __builtin_amdgcn_wmma_f32_16x16x32_f16(
            false, af[s], false, bf[q], (short)0, acc[q][s], false, false);
  }

  // Epilogue: relu + mean over 8 consecutive oc. D-matrix layout puts exactly one
  // pooled group's 8 rows in the 8 accumulator VGPRs of each lane -> in-register pool.
  const int chalf = (lane < 16) ? 0 : 8;
#pragma unroll
  for (int q = 0; q < 2; ++q) {
    const int pos = n_base + wn * 32 + q * 16 + nf;
    const int b = pos / IMG, rem = pos % IMG;
#pragma unroll
    for (int s = 0; s < 4; ++s) {
      float sum = 0.f;
#pragma unroll
      for (int v = 0; v < 8; ++v) sum += fmaxf(acc[q][s][v], 0.f);
      const int c = (oc_base + mrow + s * 16 + chalf) >> 3;  // pooled channel 0..31
      p[(size_t)b * PELEM + c * IMG + rem] = sum * 0.125f;
    }
  }
}

// --------------- kernel 5: fully connected head ---------------------------------------
__global__ void fc_kernel(const float* __restrict__ p, const float* __restrict__ fw,
                          const float* __restrict__ fb, float* __restrict__ out) {
  __shared__ float red[NCLS][256];
  const int b = blockIdx.x, tid = threadIdx.x;
  float a[NCLS];
#pragma unroll
  for (int j = 0; j < NCLS; ++j) a[j] = 0.f;
  const float* pb = p + (size_t)b * PELEM;
  for (int i = tid; i < PELEM; i += 256) {
    float v = pb[i];
#pragma unroll
    for (int j = 0; j < NCLS; ++j) a[j] += v * fw[(size_t)j * PELEM + i];
  }
#pragma unroll
  for (int j = 0; j < NCLS; ++j) red[j][tid] = a[j];
  __syncthreads();
  for (int off = 128; off > 0; off >>= 1) {
    if (tid < off) {
#pragma unroll
      for (int j = 0; j < NCLS; ++j) red[j][tid] += red[j][tid + off];
    }
    __syncthreads();
  }
  if (tid < NCLS) out[b * NCLS + tid] = red[tid][0] + fb[tid];
}

// --------------- launch ---------------------------------------------------------------
extern "C" void kernel_launch(void* const* d_in, const int* in_sizes, int n_in,
                              void* d_out, int out_size, void* d_ws, size_t ws_size,
                              hipStream_t stream) {
  (void)in_sizes; (void)n_in; (void)out_size; (void)ws_size;
  const float* x  = (const float*)d_in[0];
  const float* bw = (const float*)d_in[1];
  const float* w2 = (const float*)d_in[2];
  const float* fw = (const float*)d_in[3];
  const float* fb = (const float*)d_in[4];
  float* out = (float*)d_out;

  char* ws = (char*)d_ws;
  float*    w1f = (float*)(ws + 0);                    //   4,608 B
  _Float16* wp  = (_Float16*)(ws + 8192);              // 589,824 B  [tap][oc][ic] f16
  _Float16* h   = (_Float16*)(ws + (1ull << 20));      // 102.8 MB   [ic][b*784+pos] f16
  float*    p   = (float*)(ws + 109051904ull);         //  51.4 MB   pooled activations

  rot_filters<<<dim3((1152 + 255) / 256), 256, 0, stream>>>(bw, w1f);
  pack_w2<<<dim3(294912 / 256), 256, 0, stream>>>(w2, wp);
  conv1_relu<<<dim3(NPOS / 256, C1OUT), 256, 0, stream>>>(x, w1f, h);
  conv2_wmma_pool<<<dim3(NPOS / 128, 2), 256, 0, stream>>>(h, wp, p);
  fc_kernel<<<512, 256, 0, stream>>>(p, fw, fb, out);
}
